// DCConv_49606872269284
// MI455X (gfx1250) — matrix operand order repaired and checked by reading, not compiled, not run on hardware
//
#include <hip/hip_runtime.h>
#include <math.h>

typedef __attribute__((ext_vector_type(16))) _Float16 v16h;
typedef __attribute__((ext_vector_type(8)))  float    v8f;

union F16x16 { v16h v; unsigned u[8]; };

#define NPTS   131072          // B*N*K
#define NPTS1  4096            // B*N
#define EPSF   1e-5f

// stats region offsets (floats, from ws base)
#define ST_HM   0     // 10: sum h[i]
#define ST_HH   16    // 55: sum h[i]*h[j], packed upper-tri
#define ST_DCS  80    // 64: dc sum
#define ST_DCQ  144   // 64: dc sumsq
#define ST_B1S  208   // 128: bn1 sum
#define ST_B1Q  336   // 128: bn1 sumsq
// params region offsets (floats, from ws+512)
#define PAR_SNS 0
#define PAR_SNT 8
#define PAR_HS  16
#define PAR_HT  48
#define PAR_DCS 80
#define PAR_DCT 144
#define PAR_B1S 208
#define PAR_B1T 336

__device__ __forceinline__ void compute_h(const float* __restrict__ in,
                                          int b, int n, int k, float* h) {
  int idx  = b * 67 * 32768 + n * 32 + k;   // ch stride = 32768
  int idx0 = b * 67 * 32768 + n * 32;       // k = 0
  float a0 = in[idx + 0 * 32768], a1 = in[idx + 1 * 32768], a2 = in[idx + 2 * 32768];
  float d0 = in[idx + 3 * 32768], d1 = in[idx + 4 * 32768], d2 = in[idx + 5 * 32768];
  float x0 = in[idx0 + 0 * 32768], x1 = in[idx0 + 1 * 32768], x2 = in[idx0 + 2 * 32768];
  h[0] = sqrtf(d0 * d0 + d1 * d1 + d2 * d2);
  h[1] = x0; h[2] = x1; h[3] = x2;
  h[4] = a0; h[5] = a1; h[6] = a2;
  h[7] = d0; h[8] = d1; h[9] = d2;
}

// ---------------- K0: zero stats ----------------
__global__ void k0_zero(float* stats) {
  int t = threadIdx.x;
  stats[t] = 0.f;
  stats[t + 256] = 0.f;
}

// ---------------- K1: h moments ----------------
__global__ __launch_bounds__(256) void k1_moments(const float* __restrict__ in,
                                                  float* __restrict__ stats) {
  int t = blockIdx.x * 256 + threadIdx.x;   // exactly NPTS threads
  int k = t & 31, n = (t >> 5) & 1023, b = t >> 15;
  float h[10];
  compute_h(in, b, n, k, h);
  float acc[65];
  int c = 0;
#pragma unroll
  for (int i = 0; i < 10; ++i) acc[c++] = h[i];
#pragma unroll
  for (int i = 0; i < 10; ++i)
#pragma unroll
    for (int j = i; j < 10; ++j) acc[c++] = h[i] * h[j];
#pragma unroll
  for (int e = 0; e < 65; ++e) {
#pragma unroll
    for (int off = 16; off >= 1; off >>= 1) acc[e] += __shfl_xor(acc[e], off);
  }
  if ((threadIdx.x & 31) == 0) {
#pragma unroll
    for (int e = 0; e < 65; ++e) {
      int dst = (e < 10) ? (ST_HM + e) : (ST_HH + (e - 10));
      atomicAdd(&stats[dst], acc[e]);
    }
  }
}

// ---------------- K2: finalize sn/h BN from moments ----------------
__global__ void k2_fin_pre(const float* __restrict__ stats,
                           const float* sn_w1, const float* sn_g1, const float* sn_b1,
                           const float* h_w1, const float* h_g1, const float* h_b1,
                           float* __restrict__ prm) {
  int t = threadIdx.x;
  if (t >= 40) return;
  const float inv = 1.f / (float)NPTS;
  float Eh[10];
#pragma unroll
  for (int i = 0; i < 10; ++i) Eh[i] = stats[ST_HM + i] * inv;
  const float* w; float g, bb; int po, pt;
  if (t < 8) { w = sn_w1 + t * 10; g = sn_g1[t]; bb = sn_b1[t]; po = PAR_SNS + t; pt = PAR_SNT + t; }
  else { int cc = t - 8; w = h_w1 + cc * 10; g = h_g1[cc]; bb = h_b1[cc]; po = PAR_HS + cc; pt = PAR_HT + cc; }
  float m = 0.f;
#pragma unroll
  for (int i = 0; i < 10; ++i) m += w[i] * Eh[i];
  float ex2 = 0.f;
  for (int i = 0; i < 10; ++i)
    for (int j = 0; j < 10; ++j) {
      int ii = (i < j) ? i : j, jj = (i < j) ? j : i;
      int pidx = ST_HH + ii * 10 - (ii * (ii - 1)) / 2 + (jj - ii);
      ex2 += w[i] * w[j] * stats[pidx] * inv;
    }
  float var = ex2 - m * m;
  float s = g * rsqrtf(var + EPSF);
  prm[po] = s;
  prm[pt] = bb - m * s;
}

// ---------------- K3: fused main kernel (scores, d1, WMMA GEMM, pow, contract) ----
__global__ __launch_bounds__(128) void k3_main(
    const float* __restrict__ in, const float* __restrict__ pclip,
    const float* __restrict__ sn_w1, const float* __restrict__ sn_w2,
    const float* __restrict__ sn_b2, const float* __restrict__ h_w1,
    const float* __restrict__ h_w2, const float* __restrict__ h_b2,
    const float* __restrict__ prm, float* __restrict__ agg,
    float* __restrict__ stats) {
  __shared__ _Float16 wlds[512 * 32];   // h_w2 as f16, 32 KB
  __shared__ float    hb2s[512];
  __shared__ _Float16 d1s[4][512];      // 16 pts x 32 ch per wave
  __shared__ float    scs[4][128];      // 16 pts x 8 scores per wave
  __shared__ float    dst[128];         // dc sum[64] | sumsq[64]

  const int tid = threadIdx.x;
  // stage weights f32 -> f16
  for (int i = tid; i < 512 * 32; i += 128) wlds[i] = (_Float16)h_w2[i];
  for (int i = tid; i < 512; i += 128) hb2s[i] = h_b2[i];
  if (tid < 128) dst[tid] = 0.f;
  __syncthreads();

  const int w = tid >> 5, lane = tid & 31;
  const int g = blockIdx.x * 4 + w;
  const int p_ = lane & 15;
  const int point = g * 16 + p_;
  const int k = point & 31, n = (point >> 5) & 1023, b = point >> 15;
  const float pc = fminf(fmaxf(pclip[0], 1.f), 2.f);

  float h[10];
  compute_h(in, b, n, k, h);

  if (lane < 16) {
    // attention branch: s1 -> s2 -> softmax
    float s1v[8];
#pragma unroll
    for (int j = 0; j < 8; ++j) {
      float sp = 0.f;
#pragma unroll
      for (int i = 0; i < 10; ++i) sp += sn_w1[j * 10 + i] * h[i];
      s1v[j] = fmaxf(sp * prm[PAR_SNS + j] + prm[PAR_SNT + j], 0.f);
    }
    float s2v[8], mx = -1e30f;
#pragma unroll
    for (int m = 0; m < 8; ++m) {
      float v = sn_b2[m];
#pragma unroll
      for (int j = 0; j < 8; ++j) v += sn_w2[m * 8 + j] * s1v[j];
      s2v[m] = v; mx = fmaxf(mx, v);
    }
    float se = 0.f;
#pragma unroll
    for (int m = 0; m < 8; ++m) { s2v[m] = __expf(s2v[m] - mx); se += s2v[m]; }
    float is = 1.f / se;
#pragma unroll
    for (int m = 0; m < 8; ++m) scs[w][p_ * 8 + m] = s2v[m] * is;
  }
  // d1: lanes 0-15 -> c 0..15, lanes 16-31 -> c 16..31 (same point p_)
  {
    int c0 = (lane < 16) ? 0 : 16;
#pragma unroll
    for (int cc = 0; cc < 16; ++cc) {
      int c = c0 + cc;
      float dp = 0.f;
#pragma unroll
      for (int i = 0; i < 10; ++i) dp += h_w1[c * 10 + i] * h[i];
      float d1v = fmaxf(dp * prm[PAR_HS + c] + prm[PAR_HT + c], 0.f);
      d1s[w][p_ * 32 + c] = (_Float16)d1v;
    }
  }
  __syncthreads();

  // B fragment: column = point p_, K = d1 channels
  F16x16 bfr;
  {
    const unsigned* brow = (const unsigned*)(&d1s[w][p_ * 32]);
    int cb2 = (lane < 16) ? 0 : 4;
#pragma unroll
    for (int v = 0; v < 8; ++v) bfr.u[v] = brow[cb2 + (v & 3) + (v >> 2) * 8];
  }
  // preload x values this lane needs
  const int coff = (lane < 16) ? 0 : 8;
  float xv[32];
  {
    int base = b * 67 * 32768 + n * 32 + k;
#pragma unroll
    for (int cg = 0; cg < 4; ++cg)
#pragma unroll
      for (int r = 0; r < 8; ++r) {
        int c = cg * 16 + coff + r;
        xv[cg * 8 + r] = in[base + (3 + c) * 32768];
      }
  }
  float aggl[32];
#pragma unroll
  for (int e = 0; e < 32; ++e) aggl[e] = 0.f;

  const int M = lane & 15;
  const int kb2 = (lane < 16) ? 0 : 4;
  const int rsel = (lane < 16) ? 0 : 8;
  // m loop dynamic, cg loop fully unrolled -> all register-array indices constant
  for (int m = 0; m < 8; ++m) {
    float sm = scs[w][p_ * 8 + m];
    const unsigned* arow0 = ((const unsigned*)wlds) + (m * 64 + M) * 16;
#pragma unroll
    for (int cg = 0; cg < 4; ++cg) {
      F16x16 afr;
      const unsigned* arow = arow0 + cg * 256;   // +16 rows per cg
#pragma unroll
      for (int v = 0; v < 4; ++v) { afr.u[v] = arow[kb2 + v]; afr.u[v + 4] = arow[kb2 + 8 + v]; }
      v8f cz = {};
      v8f acc = __builtin_amdgcn_wmma_f32_16x16x32_f16(
          false, afr.v, false, bfr.v, (short)0, cz, false, false);
      // epilogue: d2 -> y -> signed pow (native exp/log) -> weighted accumulate
#pragma unroll
      for (int r = 0; r < 8; ++r) {
        int o = (m * 4 + cg) * 16 + rsel + r;
        float d2v = acc[r] + hb2s[o];
        float y = d2v * xv[cg * 8 + r];
        float sgn = (y > 0.f) ? 1.f : ((y < 0.f) ? -1.f : 0.f);
        float base_ = fabsf(y + 1e-6f) + 1e-6f;   // strictly positive
        float z = __expf(pc * __logf(base_)) * sgn;
        aggl[cg * 8 + r] += sm * z;
      }
    }
  }
  // store agg (B,64,N,K) + dc stats
#pragma unroll
  for (int e = 0; e < 32; ++e) {
    int cg = e >> 3, r = e & 7;
    int c = cg * 16 + coff + r;
    agg[b * 2097152 + c * 32768 + n * 32 + k] = aggl[e];
  }
#pragma unroll
  for (int e = 0; e < 32; ++e) {
    float s = aggl[e], q = aggl[e] * aggl[e];
#pragma unroll
    for (int off = 8; off >= 1; off >>= 1) { s += __shfl_xor(s, off); q += __shfl_xor(q, off); }
    if (p_ == 0) {
      int cg = e >> 3, r = e & 7;
      int c = cg * 16 + coff + r;
      atomicAdd(&dst[c], s);
      atomicAdd(&dst[64 + c], q);
    }
  }
  __syncthreads();
  if (tid < 64) atomicAdd(&stats[ST_DCS + tid], dst[tid]);
  else if (tid < 128) atomicAdd(&stats[ST_DCQ + (tid - 64)], dst[tid]);
}

// ---------------- K4: finalize dc BN ----------------
__global__ void k4_fin_dc(const float* stats, const float* dc_g, const float* dc_b,
                          float* prm) {
  int t = threadIdx.x;
  if (t >= 64) return;
  const float inv = 1.f / (float)NPTS;
  float m = stats[ST_DCS + t] * inv;
  float v = stats[ST_DCQ + t] * inv - m * m;
  float s = dc_g[t] * rsqrtf(v + EPSF);
  prm[PAR_DCS + t] = s;
  prm[PAR_DCT + t] = dc_b[t] - m * s;
}

// ---------------- K5: BN + relu + max over K ----------------
__global__ __launch_bounds__(256) void k5_max(const float* __restrict__ agg,
                                              const float* __restrict__ prm,
                                              float* __restrict__ out1) {
  int t = blockIdx.x * 256 + threadIdx.x;   // 262144 threads
  int n = t & 1023, c = (t >> 10) & 63, b = t >> 16;
  float s = prm[PAR_DCS + c], tt = prm[PAR_DCT + c];
  const float4* row = (const float4*)(agg + (size_t)b * 2097152 + c * 32768 + n * 32);
  float mx = -1e30f;
#pragma unroll
  for (int i = 0; i < 8; ++i) {
    float4 v = row[i];
    mx = fmaxf(mx, fmaxf(fmaxf(v.x * s + tt, v.y * s + tt),
                         fmaxf(v.z * s + tt, v.w * s + tt)));
  }
  out1[(b * 64 + c) * 1024 + n] = fmaxf(mx, 0.f);
}

// ---------------- K6: cr_w GEMM (WMMA) + bn1 stats ----------------
__global__ __launch_bounds__(128) void k6_cr(const float* __restrict__ out1,
                                             const float* __restrict__ cr_w,
                                             const float* __restrict__ cr_b,
                                             float* __restrict__ out2,
                                             float* __restrict__ stats) {
  __shared__ _Float16 wl[128 * 64];    // cr_w f16, 16 KB
  __shared__ _Float16 b1s[4][16 * 64]; // 16 pts x 64 ch per wave
  __shared__ float    st[256];

  const int tid = threadIdx.x;
  for (int i = tid; i < 128 * 64; i += 128) wl[i] = (_Float16)cr_w[i];
  st[tid] = 0.f; st[tid + 128] = 0.f;
  __syncthreads();

  const int w = tid >> 5, lane = tid & 31;
  const int g = blockIdx.x * 4 + w;
  const int p_ = lane & 15;
  const int pt = g * 16 + p_;
  const int b = pt >> 10, n = pt & 1023;

  // stage 16 points x 64 channels of out1 as f16
  {
    int cbase = (lane < 16) ? 0 : 32;
#pragma unroll
    for (int cc = 0; cc < 32; ++cc) {
      int c = cbase + cc;
      b1s[w][p_ * 64 + c] = (_Float16)out1[(b * 64 + c) * 1024 + n];
    }
  }
  __syncthreads();

  const int M = lane & 15;
  const int rsel = (lane < 16) ? 0 : 8;
  for (int t = 0; t < 8; ++t) {
    v8f acc = {};
#pragma unroll
    for (int kc = 0; kc < 2; ++kc) {
      F16x16 afr, bfr;
      const unsigned* arow = ((const unsigned*)wl) + (t * 16 + M) * 32;
      int kb2 = kc * 16 + ((lane < 16) ? 0 : 4);
#pragma unroll
      for (int v = 0; v < 4; ++v) { afr.u[v] = arow[kb2 + v]; afr.u[v + 4] = arow[kb2 + 8 + v]; }
      const unsigned* brow = (const unsigned*)(&b1s[w][p_ * 64]);
      int cb2 = kc * 16 + ((lane < 16) ? 0 : 4);
#pragma unroll
      for (int v = 0; v < 8; ++v) bfr.u[v] = brow[cb2 + (v & 3) + (v >> 2) * 8];
      acc = __builtin_amdgcn_wmma_f32_16x16x32_f16(
          false, afr.v, false, bfr.v, (short)0, acc, false, false);
    }
#pragma unroll
    for (int r = 0; r < 8; ++r) {
      int o = t * 16 + rsel + r;
      float val = acc[r] + cr_b[o];
      out2[(b * 128 + o) * 1024 + n] = val;
      float s = val, q = val * val;
#pragma unroll
      for (int off = 8; off >= 1; off >>= 1) { s += __shfl_xor(s, off); q += __shfl_xor(q, off); }
      if (p_ == 0) { atomicAdd(&st[o], s); atomicAdd(&st[128 + o], q); }
    }
  }
  __syncthreads();
  atomicAdd(&stats[ST_B1S + tid], st[tid]);
  atomicAdd(&stats[ST_B1Q + tid], st[tid + 128]);
}

// ---------------- K7: finalize bn1 ----------------
__global__ void k7_fin_b1(const float* stats, const float* bn1_g, const float* bn1_b,
                          float* prm) {
  int t = threadIdx.x;
  const float inv = 1.f / (float)NPTS1;
  float m = stats[ST_B1S + t] * inv;
  float v = stats[ST_B1Q + t] * inv - m * m;
  float s = bn1_g[t] * rsqrtf(v + EPSF);
  prm[PAR_B1S + t] = s;
  prm[PAR_B1T + t] = bn1_b[t] - m * s;
}

// ---------------- K8: apply bn1 + relu ----------------
__global__ __launch_bounds__(256) void k8_apply(const float* __restrict__ out2,
                                                const float* __restrict__ prm,
                                                float* __restrict__ out) {
  int t = blockIdx.x * 256 + threadIdx.x;  // 524288
  int o = (t >> 10) & 127;
  out[t] = fmaxf(out2[t] * prm[PAR_B1S + o] + prm[PAR_B1T + o], 0.f);
}

extern "C" void kernel_launch(void* const* d_in, const int* in_sizes, int n_in,
                              void* d_out, int out_size, void* d_ws, size_t ws_size,
                              hipStream_t stream) {
  const float* in    = (const float*)d_in[0];
  const float* p     = (const float*)d_in[1];
  const float* sn_w1 = (const float*)d_in[2];
  const float* sn_g1 = (const float*)d_in[3];
  const float* sn_b1 = (const float*)d_in[4];
  const float* sn_w2 = (const float*)d_in[5];
  const float* sn_b2 = (const float*)d_in[6];
  const float* h_w1  = (const float*)d_in[7];
  const float* h_g1  = (const float*)d_in[8];
  const float* h_b1  = (const float*)d_in[9];
  const float* h_w2  = (const float*)d_in[10];
  const float* h_b2  = (const float*)d_in[11];
  const float* dc_g  = (const float*)d_in[12];
  const float* dc_b  = (const float*)d_in[13];
  const float* cr_w  = (const float*)d_in[14];
  const float* cr_b  = (const float*)d_in[15];
  const float* bn1_g = (const float*)d_in[16];
  const float* bn1_b = (const float*)d_in[17];

  float* wsf   = (float*)d_ws;
  float* stats = wsf;
  float* prm   = wsf + 512;
  float* agg   = wsf + 4096;            // B*64*N*K f32 = 33.5 MB
  float* out1  = agg + 8388608;         // B*64*N
  float* out2  = out1 + 262144;         // B*128*N
  float* outp  = (float*)d_out;

  k0_zero<<<1, 256, 0, stream>>>(stats);
  k1_moments<<<512, 256, 0, stream>>>(in, stats);
  k2_fin_pre<<<1, 64, 0, stream>>>(stats, sn_w1, sn_g1, sn_b1, h_w1, h_g1, h_b1, prm);
  k3_main<<<2048, 128, 0, stream>>>(in, p, sn_w1, sn_w2, sn_b2, h_w1, h_w2, h_b2,
                                    prm, agg, stats);
  k4_fin_dc<<<1, 64, 0, stream>>>(stats, dc_g, dc_b, prm);
  k5_max<<<1024, 256, 0, stream>>>(agg, prm, out1);
  k6_cr<<<64, 128, 0, stream>>>(out1, cr_w, cr_b, out2, stats);
  k7_fin_b1<<<1, 128, 0, stream>>>(stats, bn1_g, bn1_b, prm);
  k8_apply<<<2048, 256, 0, stream>>>(out2, prm, outp);
}
